// MMHeadDecoder_8100308320787
// MI455X (gfx1250) — matrix operand order repaired and checked by reading, not compiled, not run on hardware
//
#include <hip/hip_runtime.h>
#include <hip/hip_bf16.h>

// ---------------------------------------------------------------------------
// MMHeadDecoder for MI455X (gfx1250): bf16 WMMA pipeline
//   gather -> [ L x (selfattn, xattn, ffn) ] -> decoder
// All GEMMs via v_wmma_f32_16x16x32_bf16 (wave32, 16x16 tiles, K-step 32).
// Attention V-staging uses gfx1250 async global->LDS copies (ASYNCcnt).
// ---------------------------------------------------------------------------

typedef __bf16 bf16;
typedef __attribute__((ext_vector_type(16))) __bf16 v16bf;
typedef __attribute__((ext_vector_type(8)))  float  v8f;

#define NTOK   65536
#define DMODEL 128
#define NCHUNK 128
#define SEQ    512
#define NHEAD  4
#define DHEAD  32
#define FFDIM  2048
#define NLAYER 4

union AFrag { v16bf v; unsigned int u[8]; };

__device__ __forceinline__ v8f wmma_bf16(v16bf a, v16bf b, v8f c) {
  return __builtin_amdgcn_wmma_f32_16x16x32_bf16(false, a, false, b, (short)0, c,
                                                 false, false);
}

// A fragment: 16(M) x 32(K), row-major source, row stride ld (elements).
// Layout: lanes 0-15 rows 0-15 hold K=0..15;
//         lanes 16-31 same rows hold K=16..31 (matches load_frag_b).
__device__ __forceinline__ v16bf load_frag_a(const bf16* base, int ld) {
  int lane = threadIdx.x & 31;
  int m    = lane & 15;
  int h    = (lane & 16) ? 8 : 0;          // uint-pair offset (k/2)
  const unsigned int* p = (const unsigned int*)(base + (size_t)m * ld);
  AFrag f;
#pragma unroll
  for (int j = 0; j < 4; ++j) f.u[j]     = p[h + j];
#pragma unroll
  for (int j = 0; j < 4; ++j) f.u[4 + j] = p[4 + h + j];
  return f.v;
}

// B fragment: 32(K) x 16(N). Source is "N-major": row n holds K contiguous
// (i.e. W[n][k] row-major weights, or a transposed activation tile).
// Contiguous k order (as load_frag_a): lanes 0-15 hold K=0..15 for N=lane, lanes 16-31 hold K=16..31.
__device__ __forceinline__ v16bf load_frag_b(const bf16* base, int ld) {
  int lane = threadIdx.x & 31;
  int n    = lane & 15;
  int kb   = (lane & 16) ? 16 : 0;
  const unsigned int* p = (const unsigned int*)(base + (size_t)n * ld + kb);
  AFrag f;
#pragma unroll
  for (int j = 0; j < 8; ++j) f.u[j] = p[j];
  return f.v;
}

// ---------------------------------------------------------------------------
// Front-end kernels
// ---------------------------------------------------------------------------

__global__ __launch_bounds__(256) void k_gather(const float* __restrict__ before,
                                                const float* __restrict__ after,
                                                const int* __restrict__ vc,
                                                float* __restrict__ x,
                                                bf16* __restrict__ xb) {
  size_t id = (size_t)blockIdx.x * 256 + threadIdx.x;   // NTOK*128 total
  int n = (int)(id >> 7), c = (int)(id & 127);
  int y = vc[n * 3 + 1], xx = vc[n * 3 + 2];
  const float* src = (c < 64) ? before : after;
  float v = src[(((size_t)(c & 63)) * 512 + y) * 512 + xx];
  x[id]  = v;
  xb[id] = (bf16)v;
}

__global__ __launch_bounds__(256) void k_mem(const float* __restrict__ po,
                                             const float* __restrict__ off_w,
                                             const float* __restrict__ off_b,
                                             bf16* __restrict__ memb) {
  size_t id = (size_t)blockIdx.x * 256 + threadIdx.x;   // NTOK*128 total
  int n = (int)(id >> 7), d = (int)(id & 127);
  float v = off_b[d] + po[n * 3 + 0] * off_w[d * 3 + 0]
                     + po[n * 3 + 1] * off_w[d * 3 + 1]
                     + po[n * 3 + 2] * off_w[d * 3 + 2];
  memb[id] = (bf16)v;
}

__global__ __launch_bounds__(256) void k_cvt(const float* __restrict__ src,
                                             bf16* __restrict__ dst) {
  size_t id = (size_t)blockIdx.x * 256 + threadIdx.x;
  dst[id] = (bf16)src[id];
}

// ---------------------------------------------------------------------------
// Generic GEMM, K=128:  out[M x N](bf16) = A[M x 128] @ W[N x 128]^T + bias
// grid = (M/16, N/128); 256 threads = 8 waves, one 16x16 tile per wave.
// ---------------------------------------------------------------------------
__global__ __launch_bounds__(256) void k_gemm128(const bf16* __restrict__ A,
                                                 const bf16* __restrict__ W,
                                                 const float* __restrict__ bias,
                                                 bf16* __restrict__ out,
                                                 int ldout) {
  int wave = threadIdx.x >> 5;
  int mt = blockIdx.x;
  int nt = blockIdx.y * 8 + wave;
  const bf16* a0 = A + (size_t)mt * 16 * 128;
  const bf16* w0 = W + (size_t)nt * 16 * 128;
  v8f c = {};
#pragma unroll
  for (int k0 = 0; k0 < 128; k0 += 32)
    c = wmma_bf16(load_frag_a(a0 + k0, 128), load_frag_b(w0 + k0, 128), c);
  int lane = threadIdx.x & 31;
  int n = lane & 15, mo = (lane & 16) ? 8 : 0;
  float bv = bias[nt * 16 + n];
#pragma unroll
  for (int r = 0; r < 8; ++r)
    out[(size_t)(mt * 16 + mo + r) * ldout + nt * 16 + n] = (bf16)(c[r] + bv);
}

// ---------------------------------------------------------------------------
// Flash attention per (chunk, head). qkv layout: [token][384] bf16,
// q at col 0, k at 128, v at 256, each head h at +h*32.
// grid = NCHUNK*NHEAD blocks, 256 threads (8 waves); each wave owns 4 Q-tiles.
// V is staged through LDS with gfx1250 async global->LDS B128 copies, then
// transposed in LDS so PV B-fragments load as contiguous dwords.
// ---------------------------------------------------------------------------
__global__ __launch_bounds__(256) void k_attn(const bf16* __restrict__ qkv,
                                              bf16* __restrict__ attn_o) {
  __shared__ __align__(16) bf16 Vt[DHEAD * SEQ];   // 32 KB, V transposed
  __shared__ __align__(16) bf16 Pst[8 * 16 * 32];  // 8 KB: async bounce, then P staging
  int chunk = blockIdx.x >> 2;
  int h     = blockIdx.x & 3;
  const bf16* base = qkv + (size_t)chunk * SEQ * 384;

  // ---- stage V: 4 chunks of 128 rows; each row is 64B contiguous ----
  unsigned pstBase = (unsigned)(unsigned long long)(size_t)&Pst[0];
  for (int cch = 0; cch < 4; ++cch) {
    int r0 = cch * 128;
    for (int t = threadIdx.x; t < 512; t += 256) {   // 2 per thread, uniform
      int lr = t >> 2, seg = t & 3;
      unsigned lds = pstBase + lr * 64 + seg * 16;
      unsigned long long g =
          (unsigned long long)(size_t)(const void*)
              (base + (size_t)(r0 + lr) * 384 + 256 + h * 32) +
          (unsigned long long)(seg * 16);
      asm volatile("global_load_async_to_lds_b128 %0, %1, off"
                   :: "v"(lds), "v"(g) : "memory");
    }
    asm volatile("s_wait_asynccnt 0" ::: "memory");  // own-wave async done
    __syncthreads();                                  // all waves' data in LDS
    const bf16* pb = Pst;
    for (int i = threadIdx.x; i < 128 * 32; i += 256) {
      int lr = i >> 5, d = i & 31;
      Vt[d * SEQ + r0 + lr] = pb[lr * 32 + d];
    }
    __syncthreads();                                  // before Pst is reused
  }

  int wave = threadIdx.x >> 5;
  int lane = threadIdx.x & 31;
  int n = lane & 15, mo = (lane & 16) ? 8 : 0;
  bf16* pst = Pst + wave * 512;
  const float scale = 0.17677669529663687f;  // 1/sqrt(32)

  for (int qt = wave; qt < SEQ / 16; qt += 8) {
    v16bf aq = load_frag_a(base + (size_t)qt * 16 * 384 + h * 32, 384);
    float mrun[8], lrun[8];
#pragma unroll
    for (int r = 0; r < 8; ++r) { mrun[r] = -3.0e38f; lrun[r] = 0.0f; }
    v8f o0 = {}, o1 = {};

    for (int kt = 0; kt < SEQ / 32; ++kt) {        // 32 keys per step
      int key0 = kt * 32;
      const bf16* kb0 = base + (size_t)key0 * 384 + 128 + h * 32;
      v16bf b0 = load_frag_b(kb0, 384);
      v16bf b1 = load_frag_b(kb0 + 16 * 384, 384);
      v8f z = {};
      v8f s0 = wmma_bf16(aq, b0, z);
      v8f s1 = wmma_bf16(aq, b1, z);
#pragma unroll
      for (int r = 0; r < 8; ++r) {
        float v0 = s0[r] * scale, v1 = s1[r] * scale;
        float mx = fmaxf(v0, v1);
#pragma unroll
        for (int off = 8; off; off >>= 1) mx = fmaxf(mx, __shfl_xor(mx, off, 16));
        float nm    = fmaxf(mrun[r], mx);
        float alpha = __expf(mrun[r] - nm);
        float p0 = __expf(v0 - nm), p1 = __expf(v1 - nm);
        float ps = p0 + p1;
#pragma unroll
        for (int off = 8; off; off >>= 1) ps += __shfl_xor(ps, off, 16);
        lrun[r] = lrun[r] * alpha + ps;
        mrun[r] = nm;
        o0[r] *= alpha;  o1[r] *= alpha;
        pst[(mo + r) * 32 + n]      = (bf16)p0;   // P tile, row-major 16x32
        pst[(mo + r) * 32 + 16 + n] = (bf16)p1;
      }
      asm volatile("s_wait_dscnt 0" ::: "memory");  // own-wave LDS RAW fence
      v16bf ap  = load_frag_a(pst, 32);
      v16bf bv0 = load_frag_b(Vt + key0, SEQ);             // dh cols 0..15
      v16bf bv1 = load_frag_b(Vt + 16 * SEQ + key0, SEQ);  // dh cols 16..31
      o0 = wmma_bf16(ap, bv0, o0);
      o1 = wmma_bf16(ap, bv1, o1);
    }
#pragma unroll
    for (int r = 0; r < 8; ++r) {
      float inv = 1.0f / lrun[r];
      size_t row = (size_t)(chunk * SEQ + qt * 16 + mo + r) * DMODEL + h * 32;
      attn_o[row + n]      = (bf16)(o0[r] * inv);
      attn_o[row + 16 + n] = (bf16)(o1[r] * inv);
    }
  }
}

// ---------------------------------------------------------------------------
// Residual + LayerNorm epilogue over a 16-row tile staged in LDS (f32).
// 256 threads: 16 rows x 16 threads, 8 cols/thread; half-wave reductions.
// ---------------------------------------------------------------------------
__device__ __forceinline__ void ln_epilogue(const float* cbuf,
                                            const float* __restrict__ g,
                                            const float* __restrict__ bt,
                                            float* __restrict__ x,
                                            bf16* __restrict__ xb, int mt) {
  int row = threadIdx.x >> 4;
  int t   = threadIdx.x & 15;
  size_t gr = (size_t)(mt * 16 + row) * 128;
  float vals[8], s = 0.0f;
#pragma unroll
  for (int j = 0; j < 8; ++j) {
    int col = t * 8 + j;
    vals[j] = x[gr + col] + cbuf[row * 128 + col];
    s += vals[j];
  }
#pragma unroll
  for (int off = 8; off; off >>= 1) s += __shfl_xor(s, off, 16);
  float mean = s * (1.0f / 128.0f), s2 = 0.0f;
#pragma unroll
  for (int j = 0; j < 8; ++j) { float d = vals[j] - mean; s2 += d * d; }
#pragma unroll
  for (int off = 8; off; off >>= 1) s2 += __shfl_xor(s2, off, 16);
  float rstd = rsqrtf(s2 * (1.0f / 128.0f) + 1e-5f);
#pragma unroll
  for (int j = 0; j < 8; ++j) {
    int col = t * 8 + j;
    float y = (vals[j] - mean) * rstd * g[col] + bt[col];
    x[gr + col]  = y;
    xb[gr + col] = (bf16)y;
  }
}

// out-projection (N=K=128) + residual + LN. grid = NTOK/16.
__global__ __launch_bounds__(256) void k_proj_ln(const bf16* __restrict__ A,
                                                 const bf16* __restrict__ W,
                                                 const float* __restrict__ bias,
                                                 const float* __restrict__ g,
                                                 const float* __restrict__ bt,
                                                 float* __restrict__ x,
                                                 bf16* __restrict__ xb) {
  __shared__ __align__(16) float cbuf[16 * 128];
  int wave = threadIdx.x >> 5;
  int mt = blockIdx.x;
  const bf16* a0 = A + (size_t)mt * 16 * 128;
  const bf16* w0 = W + (size_t)wave * 16 * 128;
  v8f c = {};
#pragma unroll
  for (int k0 = 0; k0 < 128; k0 += 32)
    c = wmma_bf16(load_frag_a(a0 + k0, 128), load_frag_b(w0 + k0, 128), c);
  int lane = threadIdx.x & 31;
  int n = lane & 15, mo = (lane & 16) ? 8 : 0;
  float bv = bias[wave * 16 + n];
#pragma unroll
  for (int r = 0; r < 8; ++r) cbuf[(mo + r) * 128 + wave * 16 + n] = c[r] + bv;
  __syncthreads();
  ln_epilogue(cbuf, g, bt, x, xb, mt);
}

// Fused FFN: 128 -> relu(2048) -> 128, hidden tile (16x2048 bf16) in LDS.
// grid = NTOK/16.
__global__ __launch_bounds__(256) void k_ffn(const bf16* __restrict__ xb_in,
                                             const bf16* __restrict__ W1,
                                             const float* __restrict__ b1,
                                             const bf16* __restrict__ W2,
                                             const float* __restrict__ b2,
                                             const float* __restrict__ g,
                                             const float* __restrict__ bt,
                                             float* __restrict__ x,
                                             bf16* __restrict__ xb) {
  __shared__ __align__(16) bf16 hbuf[16 * FFDIM];   // 64 KB
  int wave = threadIdx.x >> 5;
  int lane = threadIdx.x & 31;
  int n = lane & 15, mo = (lane & 16) ? 8 : 0;
  int mt = blockIdx.x;
  const bf16* a0 = xb_in + (size_t)mt * 16 * 128;
  // stage 1: h = relu(x @ W1^T + b1), 128 n-tiles split across 8 waves
  for (int nt = wave; nt < FFDIM / 16; nt += 8) {
    const bf16* w0 = W1 + (size_t)nt * 16 * 128;
    v8f c = {};
#pragma unroll
    for (int k0 = 0; k0 < 128; k0 += 32)
      c = wmma_bf16(load_frag_a(a0 + k0, 128), load_frag_b(w0 + k0, 128), c);
    float bv = b1[nt * 16 + n];
#pragma unroll
    for (int r = 0; r < 8; ++r) {
      float v = c[r] + bv;
      hbuf[(mo + r) * FFDIM + nt * 16 + n] = (bf16)(v > 0.0f ? v : 0.0f);
    }
  }
  __syncthreads();
  // stage 2: y = h @ W2^T + b2 (K=2048), one n-tile per wave
  const bf16* w2 = W2 + (size_t)wave * 16 * FFDIM;
  v8f c = {};
  for (int k0 = 0; k0 < FFDIM; k0 += 32) {
    __builtin_prefetch(w2 + k0 + 1024, 0, 1);  // stream next W2 tiles into L2/L0
    c = wmma_bf16(load_frag_a(hbuf + k0, FFDIM), load_frag_b(w2 + k0, FFDIM), c);
  }
  float res[8];
  float bv = b2[wave * 16 + n];
#pragma unroll
  for (int r = 0; r < 8; ++r) res[r] = c[r] + bv;
  __syncthreads();                 // done reading hbuf; reuse as f32 staging
  float* cbuf = (float*)hbuf;
#pragma unroll
  for (int r = 0; r < 8; ++r) cbuf[(mo + r) * 128 + wave * 16 + n] = res[r];
  __syncthreads();
  ln_epilogue(cbuf, g, bt, x, xb, mt);
}

// Tiny decoder MLP: gelu(x@W1^T+b1) @ W2^T + b2, per-token scalar (negligible).
__global__ __launch_bounds__(256) void k_decode(const float* __restrict__ x,
                                                const float* __restrict__ w1,
                                                const float* __restrict__ b1,
                                                const float* __restrict__ w2,
                                                const float* __restrict__ b2,
                                                float* __restrict__ out) {
  int nrow = blockIdx.x * 256 + threadIdx.x;
  const float* xr = x + (size_t)nrow * 128;
  float hbuf[32];
  for (int o = 0; o < 32; ++o) {
    float s = b1[o];
    const float* w = w1 + o * 128;
    for (int k = 0; k < 128; ++k) s += xr[k] * w[k];
    hbuf[o] = 0.5f * s * (1.0f + erff(s * 0.70710678118654752f));
  }
#pragma unroll
  for (int o = 0; o < 3; ++o) {
    float s = b2[o];
#pragma unroll
    for (int k = 0; k < 32; ++k) s += hbuf[k] * w2[o * 32 + k];
    out[(size_t)nrow * 3 + o] = s;
  }
}

// ---------------------------------------------------------------------------
// Host orchestration
// ---------------------------------------------------------------------------
extern "C" void kernel_launch(void* const* d_in, const int* in_sizes, int n_in,
                              void* d_out, int out_size, void* d_ws, size_t ws_size,
                              hipStream_t stream) {
  const float* before = (const float*)d_in[0];
  const float* after  = (const float*)d_in[1];
  const float* po     = (const float*)d_in[2];
  const int*   vc     = (const int*)d_in[3];
  const float* off_w  = (const float*)d_in[4];
  const float* off_b  = (const float*)d_in[5];
  const float* sa_in_w  = (const float*)d_in[6];
  const float* sa_in_b  = (const float*)d_in[7];
  const float* sa_out_w = (const float*)d_in[8];
  const float* sa_out_b = (const float*)d_in[9];
  const float* ca_in_w  = (const float*)d_in[10];
  const float* ca_in_b  = (const float*)d_in[11];
  const float* ca_out_w = (const float*)d_in[12];
  const float* ca_out_b = (const float*)d_in[13];
  const float* ff1_w = (const float*)d_in[14];
  const float* ff1_b = (const float*)d_in[15];
  const float* ff2_w = (const float*)d_in[16];
  const float* ff2_b = (const float*)d_in[17];
  const float* ln1_g = (const float*)d_in[18];
  const float* ln1_b = (const float*)d_in[19];
  const float* ln2_g = (const float*)d_in[20];
  const float* ln2_b = (const float*)d_in[21];
  const float* ln3_g = (const float*)d_in[22];
  const float* ln3_b = (const float*)d_in[23];
  const float* dec1_w = (const float*)d_in[24];
  const float* dec1_b = (const float*)d_in[25];
  const float* dec2_w = (const float*)d_in[26];
  const float* dec2_b = (const float*)d_in[27];

  char* ws = (char*)d_ws;
  size_t off = 0;
  auto alloc = [&](size_t bytes) -> char* {
    char* p = ws + off;
    off = (off + bytes + 255) & ~(size_t)255;
    return p;
  };
  float* x    = (float*)alloc((size_t)NTOK * DMODEL * 4);
  bf16*  xb   = (bf16*) alloc((size_t)NTOK * DMODEL * 2);
  bf16*  memb = (bf16*) alloc((size_t)NTOK * DMODEL * 2);
  bf16*  qkv  = (bf16*) alloc((size_t)NTOK * 384 * 2);
  bf16*  ao   = (bf16*) alloc((size_t)NTOK * DMODEL * 2);
  bf16*  wsa_in  = (bf16*)alloc((size_t)NLAYER * 384 * 128 * 2);
  bf16*  wsa_out = (bf16*)alloc((size_t)NLAYER * 128 * 128 * 2);
  bf16*  wca_in  = (bf16*)alloc((size_t)NLAYER * 384 * 128 * 2);
  bf16*  wca_out = (bf16*)alloc((size_t)NLAYER * 128 * 128 * 2);
  bf16*  wff1    = (bf16*)alloc((size_t)NLAYER * FFDIM * 128 * 2);
  bf16*  wff2    = (bf16*)alloc((size_t)NLAYER * 128 * FFDIM * 2);

  // front-end
  k_gather<<<NTOK * 128 / 256, 256, 0, stream>>>(before, after, vc, x, xb);
  k_mem<<<NTOK * 128 / 256, 256, 0, stream>>>(po, off_w, off_b, memb);
  k_cvt<<<NLAYER * 384 * 128 / 256, 256, 0, stream>>>(sa_in_w, wsa_in);
  k_cvt<<<NLAYER * 128 * 128 / 256, 256, 0, stream>>>(sa_out_w, wsa_out);
  k_cvt<<<NLAYER * 384 * 128 / 256, 256, 0, stream>>>(ca_in_w, wca_in);
  k_cvt<<<NLAYER * 128 * 128 / 256, 256, 0, stream>>>(ca_out_w, wca_out);
  k_cvt<<<NLAYER * FFDIM * 128 / 256, 256, 0, stream>>>(ff1_w, wff1);
  k_cvt<<<NLAYER * 128 * FFDIM / 256, 256, 0, stream>>>(ff2_w, wff2);

  const int MT = NTOK / 16;  // 4096 row tiles
  for (int i = 0; i < NLAYER; ++i) {
    const bf16* sin  = wsa_in  + (size_t)i * 384 * 128;
    const bf16* sout = wsa_out + (size_t)i * 128 * 128;
    const bf16* cin  = wca_in  + (size_t)i * 384 * 128;
    const bf16* cout = wca_out + (size_t)i * 128 * 128;
    const bf16* w1   = wff1    + (size_t)i * FFDIM * 128;
    const bf16* w2   = wff2    + (size_t)i * 128 * FFDIM;

    // --- self attention ---
    k_gemm128<<<dim3(MT, 3), 256, 0, stream>>>(xb, sin, sa_in_b + i * 384, qkv, 384);
    k_attn<<<NCHUNK * NHEAD, 256, 0, stream>>>(qkv, ao);
    k_proj_ln<<<MT, 256, 0, stream>>>(ao, sout, sa_out_b + i * 128,
                                      ln1_g + i * 128, ln1_b + i * 128, x, xb);
    // --- cross attention: q from x, k/v from mem ---
    k_gemm128<<<dim3(MT, 1), 256, 0, stream>>>(xb, cin, ca_in_b + i * 384, qkv, 384);
    k_gemm128<<<dim3(MT, 2), 256, 0, stream>>>(memb, cin + 128 * 128,
                                               ca_in_b + i * 384 + 128, qkv + 128, 384);
    k_attn<<<NCHUNK * NHEAD, 256, 0, stream>>>(qkv, ao);
    k_proj_ln<<<MT, 256, 0, stream>>>(ao, cout, ca_out_b + i * 128,
                                      ln2_g + i * 128, ln2_b + i * 128, x, xb);
    // --- FFN ---
    k_ffn<<<MT, 256, 0, stream>>>(xb, w1, ff1_b + i * FFDIM, w2, ff2_b + i * 128,
                                  ln3_g + i * 128, ln3_b + i * 128, x, xb);
  }

  k_decode<<<NTOK / 256, 256, 0, stream>>>(x, dec1_w, dec1_b, dec2_w, dec2_b,
                                           (float*)d_out);
}
